// MultiHeadAttention_10462540333621
// MI455X (gfx1250) — matrix operand order repaired and checked by reading, not compiled, not run on hardware
//
#include <hip/hip_runtime.h>

#define DIMN   1024
#define SEQ    2048
#define NHEADS 16
#define HDIM   64
#define BATCH  2

typedef __attribute__((ext_vector_type(16))) _Float16 v16h;
typedef __attribute__((ext_vector_type(8)))  float    v8f;
typedef __attribute__((ext_vector_type(4)))  int      v4i;
typedef __attribute__((ext_vector_type(8)))  int      v8i;

union H16 { v16h v; unsigned int u[8]; _Float16 h[16]; };
union F8  { v8f  v; float f[8]; };

__device__ __forceinline__ v8f wmma_f16(v16h a, v16h b, v8f c) {
  // (neg_a, A, neg_b, B, c_mod, C, reuse_a, reuse_b)
  return __builtin_amdgcn_wmma_f32_16x16x32_f16(false, a, false, b, (short)0, c,
                                                false, false);
}

// ---- Tensor Data Mover: load a 128-row x 64-halfword 2D tile into LDS ----
// D# per CDNA5 ISA (group0 128b + group1 256b, 2D tensor,
// VADDR2/VADDR3 = NULL). data_size=2B, tile_dim0=64, tile_dim1=128.
__device__ __forceinline__ void tdm_load_tile_f16(const _Float16* gptr,
                                                  unsigned lds_off,
                                                  unsigned tensor_d0,
                                                  unsigned tensor_d1) {
  unsigned long long ga = (unsigned long long)(uintptr_t)gptr;
  v4i g0;
  g0[0] = 1;                                        // count=1 (valid), user mode
  g0[1] = (int)lds_off;                             // lds_addr (bytes)
  g0[2] = (int)(unsigned)(ga & 0xffffffffu);        // global_addr[31:0]
  g0[3] = (int)((unsigned)(ga >> 32) | (2u << 30)); // global_addr[56:32] | type=2
  v8i g1;
  g1[0] = 1 << 16;                                  // data_size=1 (2 bytes)
  g1[1] = (int)((tensor_d0 & 0xffffu) << 16);       // tensor_dim0[15:0]
  g1[2] = (int)(((tensor_d0 >> 16) & 0xffffu) |     // tensor_dim0[31:16]
                ((tensor_d1 & 0xffffu) << 16));     // tensor_dim1[15:0]
  g1[3] = (int)(((tensor_d1 >> 16) & 0xffffu) |     // tensor_dim1[31:16]
                (64u << 16));                       // tile_dim0 = 64
  g1[4] = 128;                                      // tile_dim1 = 128
  g1[5] = (int)tensor_d0;                           // tensor_dim0_stride[31:0]
  g1[6] = 0;
  g1[7] = 0;
  asm volatile("tensor_load_to_lds %0, %1" :: "s"(g0), "s"(g1) : "memory");
}

// ---- fragment loaders (contiguous k = 16*(lane/16) + s per lane half, wave32) ----

// A: 16x32 f16, source row-major [*, ld]. lane m = lane&15.
__device__ __forceinline__ v16h load_frag_a(const _Float16* p, int ld, int lane) {
  int m = lane & 15, hl = lane >> 4;
  const unsigned int* q = (const unsigned int*)(p + (size_t)m * ld);
  H16 r;
#pragma unroll
  for (int v = 0; v < 8; ++v) {
    int k = (v << 1) + (hl << 4);
    r.u[v] = q[k >> 1];
  }
  return r.v;
}

// B: 32x16 f16, data stored transposed row-major Bt[n][k] (ld between n-rows).
__device__ __forceinline__ v16h load_frag_bt(const _Float16* pt, int ld, int lane) {
  int n = lane & 15, hl = lane >> 4;
  const unsigned int* q = (const unsigned int*)(pt + (size_t)n * ld + (hl << 4));
  H16 r;
#pragma unroll
  for (int v = 0; v < 8; ++v) r.u[v] = q[v];
  return r.v;
}

// B: 32x16 f16 from row-major B[k][n] (ld between k-rows).
__device__ __forceinline__ v16h load_frag_b_rm(const _Float16* p, int ld, int lane) {
  int n = lane & 15, hl = lane >> 4;
  H16 r;
#pragma unroll
  for (int j = 0; j < 16; ++j) r.h[j] = p[(size_t)(j + (hl << 4)) * ld + n];
  return r.v;
}

// ---- conversion kernels ----

__global__ __launch_bounds__(256) void k_cvt(const float* __restrict__ in,
                                             _Float16* __restrict__ out, int n) {
  int i = blockIdx.x * 256 + threadIdx.x;
  if (i < n) out[i] = (_Float16)in[i];
}

// wt[n*1024 + k] = (f16) w[k*1024 + n]
__global__ __launch_bounds__(256) void k_cvt_t(const float* __restrict__ w,
                                               _Float16* __restrict__ wt) {
  int idx = blockIdx.x * 256 + threadIdx.x;
  int n = idx >> 10, k = idx & 1023;
  wt[idx] = (_Float16)w[(size_t)k * DIMN + n];
}

// One 64-wide K-step of the 128x128 block tile: 2x(1 A-frag + 8 B-frags,
// 8 WMMAs). lA/lB must be direct __shared__ pointers so these stay ds_loads.
__device__ __forceinline__ void gemm_step(const _Float16* lA, const _Float16* lB,
                                          int w, int lane, F8 acc[8]) {
#pragma unroll
  for (int kk = 0; kk < 64; kk += 32) {
    v16h a = load_frag_a(lA + ((w << 4) << 6) + kk, 64, lane);
#pragma unroll
    for (int g = 0; g < 2; ++g) {               // 2 groups of 4 B-frags
      v16h bfr[4];
#pragma unroll
      for (int t = 0; t < 4; ++t)
        bfr[t] = load_frag_bt(lB + (((g * 4 + t) << 4) << 6) + kk, 64, lane);
#pragma unroll
      for (int t = 0; t < 4; ++t)
        acc[g * 4 + t].v = wmma_f16(a, bfr[t], acc[g * 4 + t].v);
    }
  }
}

// ---- TDM double-buffered (explicit ping-pong) 128x128 block GEMM core ----
// A [rowsA, 1024] row-major f16, Bt [1024, 1024] row-major (transposed weight).
// Block = 8 waves; wave w owns rows [m0+16w, +16), all 128 cols (8 WMMA tiles).
// Wave 0 drives the TDM; next K-tile's DMA overlaps current tile's WMMAs.
__device__ __forceinline__ void gemm_block_128(
    const _Float16* __restrict__ A, const _Float16* __restrict__ Bt,
    int m0, int n0, unsigned rowsA,
    _Float16* ldsA0, _Float16* ldsB0, _Float16* ldsA1, _Float16* ldsB1,
    F8 acc[8]) {
  int lane = threadIdx.x & 31;
  int w    = threadIdx.x >> 5;
#pragma unroll
  for (int t = 0; t < 8; ++t)
#pragma unroll
    for (int v = 0; v < 8; ++v) acc[t].f[v] = 0.f;

  unsigned offA0 = (unsigned)(uintptr_t)ldsA0;   // low 32 bits = LDS offset
  unsigned offB0 = (unsigned)(uintptr_t)ldsB0;
  unsigned offA1 = (unsigned)(uintptr_t)ldsA1;
  unsigned offB1 = (unsigned)(uintptr_t)ldsB1;

  const _Float16* Ab = A  + (size_t)m0 * DIMN;
  const _Float16* Bb = Bt + (size_t)n0 * DIMN;

  if (threadIdx.x < 32) {                        // prologue: tile 0 -> buf 0
    tdm_load_tile_f16(Ab, offA0, DIMN, rowsA);
    tdm_load_tile_f16(Bb, offB0, DIMN, DIMN);
    __builtin_amdgcn_s_wait_tensorcnt(0);
  }
  __syncthreads();

  for (int kt = 0; kt < DIMN; kt += 128) {       // ping-pong, 2 tiles per iter
    // -- phase 0: prefetch kt+64 -> buf1 (always valid: kt+64 <= 960), compute buf0
    if (threadIdx.x < 32) {
      tdm_load_tile_f16(Ab + kt + 64, offA1, DIMN, rowsA);
      tdm_load_tile_f16(Bb + kt + 64, offB1, DIMN, DIMN);
    }
    gemm_step(ldsA0, ldsB0, w, lane, acc);
    if (threadIdx.x < 32) __builtin_amdgcn_s_wait_tensorcnt(0);
    __syncthreads();                             // buf1 ready; buf0 reusable
    // -- phase 1: prefetch kt+128 -> buf0 (if any), compute buf1
    bool more = (kt + 128) < DIMN;
    if (more && threadIdx.x < 32) {
      tdm_load_tile_f16(Ab + kt + 128, offA0, DIMN, rowsA);
      tdm_load_tile_f16(Bb + kt + 128, offB0, DIMN, DIMN);
    }
    gemm_step(ldsA1, ldsB1, w, lane, acc);
    if (more) {
      if (threadIdx.x < 32) __builtin_amdgcn_s_wait_tensorcnt(0);
      __syncthreads();                           // buf0 ready; buf1 reusable
    }
  }
}

// ---- fused QKV projection: x@W -> head-split [B,H,S,64] f16 ----

__global__ __launch_bounds__(256) void k_gemm_qkv(
    const _Float16* __restrict__ xh,
    const _Float16* __restrict__ qwt, const _Float16* __restrict__ kwt,
    const _Float16* __restrict__ vwt,
    _Float16* __restrict__ Qh, _Float16* __restrict__ Kh, _Float16* __restrict__ Vh) {
  __shared__ _Float16 ldsA0[128 * 64];
  __shared__ _Float16 ldsB0[128 * 64];
  __shared__ _Float16 ldsA1[128 * 64];
  __shared__ _Float16 ldsB1[128 * 64];
  const _Float16* Bt = (blockIdx.y == 0) ? qwt : (blockIdx.y == 1) ? kwt : vwt;
  _Float16* out      = (blockIdx.y == 0) ? Qh  : (blockIdx.y == 1) ? Kh  : Vh;
  int m0 = (int)(blockIdx.x >> 3) << 7;   // 32 row blocks
  int n0 = (int)(blockIdx.x & 7) << 7;    // 8 col blocks
  F8 acc[8];
  gemm_block_128(xh, Bt, m0, n0, BATCH * SEQ, ldsA0, ldsB0, ldsA1, ldsB1, acc);

  int lane = threadIdx.x & 31, w = threadIdx.x >> 5;
  int hl = lane >> 4, n = lane & 15;
#pragma unroll
  for (int t = 0; t < 8; ++t)
#pragma unroll
    for (int v = 0; v < 8; ++v) {
      int m = m0 + (w << 4) + v + (hl << 3);
      int c = n0 + (t << 4) + n;
      int b = m >> 11, s = m & (SEQ - 1);
      int h = c >> 6,  dh = c & 63;
      out[(((size_t)(b * NHEADS + h)) << 17) + ((size_t)s << 6) + dh] =
          (_Float16)acc[t].f[v];
    }
}

// ---- flash attention: one wave per (b,h,q-tile of 16 rows) ----

__global__ __launch_bounds__(256) void k_attn(const _Float16* __restrict__ Qh,
                                              const _Float16* __restrict__ Kh,
                                              const _Float16* __restrict__ Vh,
                                              _Float16* __restrict__ ctx) {
  __shared__ _Float16 Pld[8][16][32];   // per-wave P tile (C-layout -> A-layout)
  int w    = threadIdx.x >> 5;
  int lane = threadIdx.x & 31;
  int task = blockIdx.x * 8 + w;        // 4096 = B*H*(S/16)
  int qt = task & 127;
  int bh = task >> 7;                   // b*16+h
  const _Float16* Qp = Qh + ((size_t)bh << 17);
  const _Float16* Kp = Kh + ((size_t)bh << 17);
  const _Float16* Vp = Vh + ((size_t)bh << 17);
  int q0 = qt << 4;
  int hl = lane >> 4, n = lane & 15;

  v16h aQ0 = load_frag_a(Qp + (size_t)q0 * HDIM, HDIM, lane);
  v16h aQ1 = load_frag_a(Qp + (size_t)q0 * HDIM + 32, HDIM, lane);

  float rmax[8], rsum[8];
  F8 acc[4];
#pragma unroll
  for (int v = 0; v < 8; ++v) { rmax[v] = -__builtin_inff(); rsum[v] = 0.f; }
#pragma unroll
  for (int t = 0; t < 4; ++t)
#pragma unroll
    for (int v = 0; v < 8; ++v) acc[t].f[v] = 0.f;

  int nkb = (q0 + 47) >> 5;             // causal: key blocks of 32
  for (int kb = 0; kb < nkb; ++kb) {
    int K0 = kb << 5;
    F8 s0, s1;
#pragma unroll
    for (int v = 0; v < 8; ++v) { s0.f[v] = 0.f; s1.f[v] = 0.f; }
    v16h b00 = load_frag_bt(Kp + (size_t)K0 * HDIM, HDIM, lane);
    v16h b01 = load_frag_bt(Kp + (size_t)K0 * HDIM + 32, HDIM, lane);
    s0.v = wmma_f16(aQ0, b00, s0.v);
    s0.v = wmma_f16(aQ1, b01, s0.v);
    v16h b10 = load_frag_bt(Kp + (size_t)(K0 + 16) * HDIM, HDIM, lane);
    v16h b11 = load_frag_bt(Kp + (size_t)(K0 + 16) * HDIM + 32, HDIM, lane);
    s1.v = wmma_f16(aQ0, b10, s1.v);
    s1.v = wmma_f16(aQ1, b11, s1.v);

    float nmax[8];
#pragma unroll
    for (int v = 0; v < 8; ++v) {
      int q = q0 + v + (hl << 3);
      float x0 = s0.f[v] * 0.125f;            // 1/sqrt(64)
      float x1 = s1.f[v] * 0.125f;
      if (K0 + n > q)      x0 = -__builtin_inff();
      if (K0 + 16 + n > q) x1 = -__builtin_inff();
      s0.f[v] = x0; s1.f[v] = x1;
      float mx = fmaxf(x0, x1);
#pragma unroll
      for (int d = 1; d < 16; d <<= 1) mx = fmaxf(mx, __shfl_xor(mx, d, 32));
      nmax[v] = fmaxf(rmax[v], mx);
    }
#pragma unroll
    for (int v = 0; v < 8; ++v) {
      float alpha = __expf(rmax[v] - nmax[v]);
      rmax[v] = nmax[v];
      float p0 = __expf(s0.f[v] - nmax[v]);
      float p1 = __expf(s1.f[v] - nmax[v]);
      Pld[w][v + (hl << 3)][n]      = (_Float16)p0;
      Pld[w][v + (hl << 3)][16 + n] = (_Float16)p1;
      float rs = p0 + p1;
#pragma unroll
      for (int d = 1; d < 16; d <<= 1) rs += __shfl_xor(rs, d, 32);
      rsum[v] = rsum[v] * alpha + rs;
#pragma unroll
      for (int t = 0; t < 4; ++t) acc[t].f[v] *= alpha;
    }
    // O += P @ V  (P 16x32 from LDS, V 32x64 row-major)
    v16h aP = load_frag_a(&Pld[w][0][0], 32, lane);
#pragma unroll
    for (int t = 0; t < 4; ++t) {
      v16h bv = load_frag_b_rm(Vp + (size_t)K0 * HDIM + (t << 4), HDIM, lane);
      acc[t].v = wmma_f16(aP, bv, acc[t].v);
    }
  }
  // normalize + merge heads: ctx[b*S+s][h*64 + d]
  int b = bh >> 4, h = bh & 15;
#pragma unroll
  for (int v = 0; v < 8; ++v) {
    float inv = 1.0f / rsum[v];
    size_t row = (size_t)b * SEQ + (q0 + v + (hl << 3));
#pragma unroll
    for (int t = 0; t < 4; ++t)
      ctx[row * DIMN + (h << 6) + (t << 4) + n] = (_Float16)(acc[t].f[v] * inv);
  }
}

// ---- output projection: ctx f16 @ WoT -> fp32 out ----

__global__ __launch_bounds__(256) void k_proj(const _Float16* __restrict__ ctx,
                                              const _Float16* __restrict__ wot,
                                              float* __restrict__ outp) {
  __shared__ _Float16 ldsA0[128 * 64];
  __shared__ _Float16 ldsB0[128 * 64];
  __shared__ _Float16 ldsA1[128 * 64];
  __shared__ _Float16 ldsB1[128 * 64];
  int m0 = (int)(blockIdx.x >> 3) << 7;
  int n0 = (int)(blockIdx.x & 7) << 7;
  F8 acc[8];
  gemm_block_128(ctx, wot, m0, n0, BATCH * SEQ, ldsA0, ldsB0, ldsA1, ldsB1, acc);

  int lane = threadIdx.x & 31, w = threadIdx.x >> 5;
  int hl = lane >> 4, n = lane & 15;
#pragma unroll
  for (int t = 0; t < 8; ++t)
#pragma unroll
    for (int v = 0; v < 8; ++v)
      outp[(size_t)(m0 + (w << 4) + v + (hl << 3)) * DIMN + n0 + (t << 4) + n] =
          acc[t].f[v];
}

extern "C" void kernel_launch(void* const* d_in, const int* in_sizes, int n_in,
                              void* d_out, int out_size, void* d_ws, size_t ws_size,
                              hipStream_t stream) {
  (void)in_sizes; (void)n_in; (void)out_size; (void)ws_size;
  const float* x  = (const float*)d_in[0];
  const float* Qw = (const float*)d_in[1];
  const float* Kw = (const float*)d_in[2];
  const float* Vw = (const float*)d_in[3];
  const float* Wo = (const float*)d_in[4];
  float* out = (float*)d_out;

  const size_t NX = (size_t)BATCH * SEQ * DIMN;   // 4.19M
  const size_t NW = (size_t)DIMN * DIMN;          // 1.05M
  _Float16* xh  = (_Float16*)d_ws;
  _Float16* qwt = xh  + NX;
  _Float16* kwt = qwt + NW;
  _Float16* vwt = kwt + NW;
  _Float16* wot = vwt + NW;
  _Float16* Qh  = wot + NW;
  _Float16* Kh  = Qh + NX;
  _Float16* Vh  = Kh + NX;
  _Float16* ctx = Vh + NX;                        // total ~50 MB f16 workspace

  k_cvt<<<(int)((NX + 255) / 256), 256, 0, stream>>>(x, xh, (int)NX);
  k_cvt_t<<<(int)(NW / 256), 256, 0, stream>>>(Qw, qwt);
  k_cvt_t<<<(int)(NW / 256), 256, 0, stream>>>(Kw, kwt);
  k_cvt_t<<<(int)(NW / 256), 256, 0, stream>>>(Vw, vwt);
  k_cvt_t<<<(int)(NW / 256), 256, 0, stream>>>(Wo, wot);
  // 32x8 blocks of 128x128 per projection, x3 projections
  k_gemm_qkv<<<dim3(256, 3), 256, 0, stream>>>(xh, qwt, kwt, vwt, Qh, Kh, Vh);
  // 4096 waves: B*H*(S/16)
  k_attn<<<512, 256, 0, stream>>>(Qh, Kh, Vh, ctx);
  k_proj<<<256, 256, 0, stream>>>(ctx, wot, out);
}
